// HashingMemory_63840393888332
// MI455X (gfx1250) — matrix operand order repaired and checked by reading, not compiled, not run on hardware
//
#include <hip/hip_runtime.h>
#include <math.h>

// ---- problem constants (from reference) ----
#define HEADS    4
#define K_DIM    256
#define HALF_DIM 128
#define KNN      32
#define NKEYS    512
#define D_MODEL  1024
#define BS       2048
#define NROWS    (BS * HEADS)      // 8192 query rows after head split

typedef _Float16 half8 __attribute__((ext_vector_type(8)));
typedef _Float16 v16h  __attribute__((ext_vector_type(16)));
typedef float    v8f   __attribute__((ext_vector_type(8)));

// ---------------------------------------------------------------------------
// f32 -> f16 conversion (grid-stride)
// ---------------------------------------------------------------------------
__global__ void cvt_f32_to_f16(const float* __restrict__ src,
                               _Float16* __restrict__ dst, int n) {
    for (int i = blockIdx.x * blockDim.x + threadIdx.x; i < n;
         i += gridDim.x * blockDim.x)
        dst[i] = (_Float16)src[i];
}

// ---------------------------------------------------------------------------
// WMMA fragment loaders (CDNA5 16x16x32 f16 layouts, CDNA5 ISA)
// A (16x32, MxK): lane l -> row m = l&15; halves 0..7 = K kb..kb+7,
//                 halves 8..15 = K 16+kb..16+kb+7, kb = (l>>4)*8
// ---------------------------------------------------------------------------
__device__ __forceinline__ v16h load_fragA(const _Float16* __restrict__ base,
                                           int m0, int ld, int k0, int lane) {
    int m  = m0 + (lane & 15);
    int kb = (lane >> 4) << 3;                       // 0 or 8
    const _Float16* p = base + (size_t)m * ld + k0 + kb;
    half8 lo = *(const half8*)(p);                   // K = kb .. kb+7
    half8 hi = *(const half8*)(p + 16);              // K = 16+kb .. 16+kb+7
    v16h f;
#pragma unroll
    for (int i = 0; i < 8; ++i) { f[i] = lo[i]; f[8 + i] = hi[i]; }
    return f;
}

// B (32x16, KxN), source stored row-major as src[n][k] (i.e. we compute
// q @ srcᵀ): lane l -> col n = l&15; halves 0..7 = K kc..kc+7,
// halves 8..15 = K 16+kc..16+kc+7, kc = (l>>4)*8 (same K order as A).
__device__ __forceinline__ v16h load_fragB(const _Float16* __restrict__ base,
                                           int n0, int ld, int k0, int lane) {
    int n  = n0 + (lane & 15);
    int kc = (lane >> 4) << 3;                       // 0 or 8
    const _Float16* p = base + (size_t)n * ld + k0 + kc;
    half8 lo = *(const half8*)(p);
    half8 hi = *(const half8*)(p + 16);
    v16h f;
#pragma unroll
    for (int i = 0; i < 8; ++i) { f[i] = lo[i]; f[8 + i] = hi[i]; }
    return f;
}

// ---------------------------------------------------------------------------
// Kernel 1: q = x @ Wqᵀ + b   (M=2048, N=1024, K=1024), f16 in, f32 acc,
// stores q as f16 (row-major 2048x1024 == 8192x256 after reshape).
// One wave per 16x16 tile. grid = (N/16, M/16), block = 32.
// ---------------------------------------------------------------------------
__global__ void __launch_bounds__(32)
qproj_wmma(const _Float16* __restrict__ xh, const _Float16* __restrict__ wh,
           const float* __restrict__ bias, _Float16* __restrict__ qh) {
    int n0 = blockIdx.x * 16;
    int m0 = blockIdx.y * 16;
    int lane = threadIdx.x;
    v8f c = {};
#pragma unroll 4
    for (int k0 = 0; k0 < D_MODEL; k0 += 32) {
        v16h a = load_fragA(xh, m0, D_MODEL, k0, lane);
        v16h b = load_fragB(wh, n0, D_MODEL, k0, lane);
        c = __builtin_amdgcn_wmma_f32_16x16x32_f16(false, a, false, b,
                                                   (short)0, c, false, false);
    }
    // D layout: lane l -> n = l&15, m = (l>>4)*8 + r
    int n  = n0 + (lane & 15);
    int mb = m0 + ((lane >> 4) << 3);
    float bv = bias[n];
#pragma unroll
    for (int r = 0; r < 8; ++r)
        qh[(size_t)(mb + r) * D_MODEL + n] = (_Float16)(c[r] + bv);
}

// ---------------------------------------------------------------------------
// Kernel 2: s[h] = q_h @ keysᵀ  (M=8192, N=512, K=128), h in {0,1}.
// q rows are 256 wide; half h uses columns [h*128, h*128+128).
// grid = (512/16, 8192/16, 2), block = 32.
// ---------------------------------------------------------------------------
__global__ void __launch_bounds__(32)
score_wmma(const _Float16* __restrict__ qh, const _Float16* __restrict__ kh,
           float* __restrict__ s) {
    int n0 = blockIdx.x * 16;
    int m0 = blockIdx.y * 16;
    int h  = blockIdx.z;
    int lane = threadIdx.x;
    const _Float16* qb = qh + h * HALF_DIM;          // column offset in 256-wide row
    v8f c = {};
#pragma unroll
    for (int k0 = 0; k0 < HALF_DIM; k0 += 32) {
        v16h a = load_fragA(qb, m0, K_DIM, k0, lane);
        v16h b = load_fragB(kh, n0, HALF_DIM, k0, lane);
        c = __builtin_amdgcn_wmma_f32_16x16x32_f16(false, a, false, b,
                                                   (short)0, c, false, false);
    }
    float* dst = s + (size_t)h * NROWS * NKEYS;
    int n  = n0 + (lane & 15);
    int mb = m0 + ((lane >> 4) << 3);
#pragma unroll
    for (int r = 0; r < 8; ++r)
        dst[(size_t)(mb + r) * NKEYS + n] = c[r];
}

// ---------------------------------------------------------------------------
// Kernel 3: top-32 of each 512-wide row (descending). One wave per row.
// Lane holds s[i*32 + lane], i = 0..15. 32 extraction rounds via shfl_xor.
// grid = (8192, 2), block = 32.
// ---------------------------------------------------------------------------
__global__ void __launch_bounds__(32)
topk512(const float* __restrict__ s, float* __restrict__ tv,
        int* __restrict__ ti) {
    int row = blockIdx.x;
    int h   = blockIdx.y;
    int lane = threadIdx.x;
    const float* src = s + ((size_t)h * NROWS + row) * NKEYS;
    float v[16];
#pragma unroll
    for (int i = 0; i < 16; ++i) v[i] = src[i * 32 + lane];

    float* otv = tv + ((size_t)h * NROWS + row) * KNN;
    int*   oti = ti + ((size_t)h * NROWS + row) * KNN;
    const float NEG = -__builtin_inff();

    for (int sel = 0; sel < KNN; ++sel) {
        float bv = NEG; int bi = 0x7fffffff;
#pragma unroll
        for (int i = 0; i < 16; ++i) {
            if (v[i] > bv) { bv = v[i]; bi = i * 32 + lane; }
        }
#pragma unroll
        for (int off = 16; off > 0; off >>= 1) {
            float ov = __shfl_xor(bv, off);
            int   oi = __shfl_xor(bi, off);
            if (ov > bv || (ov == bv && oi < bi)) { bv = ov; bi = oi; }
        }
        if ((bi & 31) == lane) v[bi >> 5] = NEG;     // knock out winner
        if (lane == 0) { otv[sel] = bv; oti[sel] = bi; }
    }
}

// ---------------------------------------------------------------------------
// Kernel 4: combine 32x32 candidates, top-32, fused softmax.
// Lane l owns s1 candidate i=l; s2[j] broadcast via shfl. One wave per row.
// grid = 8192, block = 32.
// ---------------------------------------------------------------------------
__global__ void __launch_bounds__(32)
combine_softmax(const float* __restrict__ tv, const int* __restrict__ ti,
                float* __restrict__ oscores, int* __restrict__ oindices) {
    int row = blockIdx.x;
    int lane = threadIdx.x;
    float a  = tv[(size_t)row * KNN + lane];
    int   ia = ti[(size_t)row * KNN + lane];
    float b  = tv[((size_t)NROWS + row) * KNN + lane];
    int   ib = ti[((size_t)NROWS + row) * KNN + lane];

    __shared__ float se[KNN];
    __shared__ int   sx[KNN];
    unsigned taken = 0u;                              // per-lane: j's consumed
    const float NEG = -__builtin_inff();
    float maxv = 0.f, sum = 0.f;

    for (int sel = 0; sel < KNN; ++sel) {
        // local best over untaken j for this lane's i
        float bv = NEG; int bj = 0;
        for (int j = 0; j < KNN; ++j) {
            float cv = a + __shfl(b, j);
            bool ok = !((taken >> j) & 1u);
            if (ok && cv > bv) { bv = cv; bj = j; }
        }
        int bi = lane;
#pragma unroll
        for (int off = 16; off > 0; off >>= 1) {
            float ov = __shfl_xor(bv, off);
            int   oi = __shfl_xor(bi, off);
            int   oj = __shfl_xor(bj, off);
            if (ov > bv || (ov == bv && oi < bi)) { bv = ov; bi = oi; bj = oj; }
        }
        if (lane == bi) taken |= (1u << bj);          // mark consumed
        int idx = __shfl(ia, bi) * NKEYS + __shfl(ib, bj);
        if (sel == 0) maxv = bv;                       // descending => first = max
        float e = expf(bv - maxv);
        sum += e;                                      // uniform across lanes
        if (lane == 0) { se[sel] = e; sx[sel] = idx; }
    }
    __syncthreads();
    oscores [(size_t)row * KNN + lane] = se[lane] / sum;
    oindices[(size_t)row * KNN + lane] = sx[lane];
}

// ---------------------------------------------------------------------------
// Kernel 5: out[b,:] = sum_k score[b,k] * values[idx[b,k],:]
// Bandwidth-bound gather: 1 GB of value rows. One block per b, float4 lanes.
// grid = 2048, block = 256.
// ---------------------------------------------------------------------------
__global__ void __launch_bounds__(256)
gather_weighted(const float* __restrict__ scores, const int* __restrict__ indices,
                const float* __restrict__ values, float* __restrict__ out) {
    int b   = blockIdx.x;
    int tid = threadIdx.x;
    __shared__ float sc[HEADS * KNN];
    __shared__ int   si[HEADS * KNN];
    if (tid < HEADS * KNN) {
        sc[tid] = scores [(size_t)b * (HEADS * KNN) + tid];
        si[tid] = indices[(size_t)b * (HEADS * KNN) + tid];
    }
    __syncthreads();
    float4 acc = make_float4(0.f, 0.f, 0.f, 0.f);
#pragma unroll 4
    for (int k = 0; k < HEADS * KNN; ++k) {
        const float4* vr =
            (const float4*)(values + (size_t)si[k] * D_MODEL);
        float4 v = vr[tid];
        float  w = sc[k];
        acc.x += w * v.x; acc.y += w * v.y; acc.z += w * v.z; acc.w += w * v.w;
    }
    ((float4*)(out + (size_t)b * D_MODEL))[tid] = acc;
}

// ---------------------------------------------------------------------------
extern "C" void kernel_launch(void* const* d_in, const int* in_sizes, int n_in,
                              void* d_out, int out_size, void* d_ws, size_t ws_size,
                              hipStream_t stream) {
    const float* x      = (const float*)d_in[0];   // (2048, 1024)
    const float* Wq     = (const float*)d_in[1];   // (1024, 1024)
    const float* bq     = (const float*)d_in[2];   // (1024,)
    const float* keys   = (const float*)d_in[3];   // (512, 128)
    const float* values = (const float*)d_in[4];   // (262144, 1024)
    float* out = (float*)d_out;                    // (2048, 1024)

    // carve workspace (256B aligned)
    char* p = (char*)d_ws;
    auto carve = [&](size_t bytes) {
        char* r = p;
        p += (bytes + 255) & ~(size_t)255;
        return r;
    };
    _Float16* xh  = (_Float16*)carve((size_t)BS * D_MODEL * 2);        // 4 MB
    _Float16* wh  = (_Float16*)carve((size_t)D_MODEL * D_MODEL * 2);   // 2 MB
    _Float16* kh  = (_Float16*)carve((size_t)NKEYS * HALF_DIM * 2);    // 128 KB
    _Float16* qh  = (_Float16*)carve((size_t)NROWS * K_DIM * 2);       // 4 MB
    float*    s   = (float*)carve((size_t)2 * NROWS * NKEYS * 4);      // 32 MB
    float*    tv  = (float*)carve((size_t)2 * NROWS * KNN * 4);        // 2 MB
    int*      ti  = (int*)  carve((size_t)2 * NROWS * KNN * 4);        // 2 MB
    float*    fsc = (float*)carve((size_t)NROWS * KNN * 4);            // 1 MB
    int*      fid = (int*)  carve((size_t)NROWS * KNN * 4);            // 1 MB
    (void)ws_size; (void)in_sizes; (void)n_in; (void)out_size;

    // 1) precision conversion
    cvt_f32_to_f16<<<4096, 256, 0, stream>>>(x,    xh, BS * D_MODEL);
    cvt_f32_to_f16<<<4096, 256, 0, stream>>>(Wq,   wh, D_MODEL * D_MODEL);
    cvt_f32_to_f16<<<256,  256, 0, stream>>>(keys, kh, NKEYS * HALF_DIM);

    // 2) q = x @ Wqᵀ + b   (WMMA)
    {
        dim3 grid(D_MODEL / 16, BS / 16);            // (64, 128)
        qproj_wmma<<<grid, 32, 0, stream>>>(xh, wh, bq, qh);
    }

    // 3) s1/s2 = q_half @ keysᵀ  (WMMA)
    {
        dim3 grid(NKEYS / 16, NROWS / 16, 2);        // (32, 512, 2)
        score_wmma<<<grid, 32, 0, stream>>>(qh, kh, s);
    }

    // 4) per-row top-32 of 512
    {
        dim3 grid(NROWS, 2);
        topk512<<<grid, 32, 0, stream>>>(s, tv, ti);
    }

    // 5) combine 32x32 -> top-32 + softmax
    combine_softmax<<<NROWS, 32, 0, stream>>>(tv, ti, fsc, fid);

    // 6) weighted value gather (bandwidth-dominant: ~1 GB @ 23.3 TB/s)
    gather_weighted<<<BS, 256, 0, stream>>>(fsc, fid, values, out);
}
